// Harmonic_14328010900085
// MI455X (gfx1250) — compile-verified, matrix operand order checked by reading, not run on hardware
//
#include <hip/hip_runtime.h>
#include <hip/hip_bf16.h>

typedef __attribute__((ext_vector_type(16))) _Float16 v16h;
typedef __attribute__((ext_vector_type(8)))  _Float16 v8h;
typedef __attribute__((ext_vector_type(8)))  float    v8f;

#define CH       128
#define NFRAMES  128
#define NHARM    16
#define NAUDIO   32768
#define VOICES   64

#define PITCH_A  136   // halfs per column row; bufA holds 128 cols
#define PITCH_B  136   // bufB holds 64 cols

// ---------------------------------------------------------------------------
// K-index pattern of a lane's 16 halfs within a 16-bit A/B WMMA fragment
// (ISA 7.12.2): two contiguous 8-runs: [hi8, hi8+7] and [16+hi8, 16+hi8+7],
// hi8 = (lane/16)*8.
// ---------------------------------------------------------------------------
__device__ __forceinline__ int frag_k(int lane, int e) {
    int v = e >> 1, p = e & 1;
    return ((v & 4) << 2) + ((lane >> 4) << 3) + ((v & 3) << 1) + p;
}

// ---------------------------------------------------------------------------
// Weight prep: ConvT weights (cin,cout,4) f32 -> per-lane WMMA A-fragment
// images f16, layout [layer][tap][kt][mtile][lane][16 halfs]  (32B per lane).
// ---------------------------------------------------------------------------
struct PrepArgs { const float* src[12]; };

__global__ __launch_bounds__(256) void prep_weights(PrepArgs p, _Float16* dst) {
    const int l = blockIdx.y;
    const int u = blockIdx.x * 256 + threadIdx.x;   // 4096 per layer
    const int lane  = u & 31;
    const int mtile = (u >> 5) & 7;
    const int kt    = (u >> 8) & 3;
    const int tap   = (u >> 10) & 3;
    const float* s = p.src[l];
    const int cout = mtile * 16 + (lane & 15);
    _Float16* d = dst + ((((size_t)l * 16 + tap * 4 + kt) * 8 + mtile) * 32 + lane) * 16;
    #pragma unroll
    for (int e = 0; e < 16; ++e) {
        const int cin = kt * 32 + frag_k(lane, e);
        d[e] = (_Float16)s[cin * 512 + cout * 4 + tap];
    }
}

// ---------------------------------------------------------------------------
// Branchless B-fragment load from LDS [col][ch] layout: two ds_load_b128,
// then cndmask-zero for invalid source columns.
// ---------------------------------------------------------------------------
__device__ __forceinline__ v16h load_bfrag(const _Float16* hin, int P, int base_k,
                                           int col, int Lin, bool ok) {
    int jc = col < 0 ? 0 : (col >= Lin ? Lin - 1 : col);
    const _Float16* hp = hin + jc * P + base_k;
    v8h lo = *(const v8h*)hp;
    v8h hi = *(const v8h*)(hp + 16);
    v16h r;
    #pragma unroll
    for (int e = 0; e < 8; ++e) {
        r[e]     = ok ? lo[e] : (_Float16)0.0f;
        r[8 + e] = ok ? hi[e] : (_Float16)0.0f;
    }
    return r;
}

// ---------------------------------------------------------------------------
// Sequence decoder: init matvec -> n_layers ConvT(k=4,s=2,p=1)+leaky via WMMA
// -> final k=3 conv to 1 channel. One block per voice, 8 waves = 8 M-tiles.
// Parity-split ConvT: y[2j] = W1 x[j] + W3 x[j-1]; y[2j+1] = W0 x[j+1] + W2 x[j].
// ---------------------------------------------------------------------------
struct SeqArgs {
    const float* init_w;          // (512,128)
    const float* init_b;          // (512)
    const float* lb[5];           // per-layer bias (128)
    const float* final_w;         // (1,128,3)
    const float* final_b;         // (1)
    const _Float16* wT;           // fragment-swizzled weights
    float* out;                   // [VOICES][NFRAMES]
    int n_layers;
    int n_frames;
};

__global__ __launch_bounds__(256) void seq_kernel(SeqArgs A, const float* x) {
    __shared__ __align__(16) _Float16 bufA[128 * PITCH_A];  // [col][ch]
    __shared__ __align__(16) _Float16 bufB[64 * PITCH_B];
    __shared__ float xs[CH];

    const int tid   = threadIdx.x;
    const int lane  = tid & 31;
    const int wave  = tid >> 5;            // M-tile (0..7)
    const int voice = blockIdx.x;

    if (tid < CH) xs[tid] = x[voice * CH + tid];
    __syncthreads();

    // Parity so the last layer (len = n_frames) lands in bufA (index 0).
    int cur = (A.n_layers & 1) ? 1 : 0;

    // ---- init conv: h[512] = init_w @ x + b, reshaped to (128 ch, 4 cols)
    {
        _Float16* h0 = cur ? bufB : bufA;
        for (int o = tid; o < 4 * CH; o += 256) {
            float acc = A.init_b[o];
            const float* wr = A.init_w + o * CH;
            for (int c = 0; c < CH; ++c) acc += wr[c] * xs[c];
            h0[(o & 3) * PITCH_A + (o >> 2)] = (_Float16)acc;  // [col][ch]
        }
    }
    __syncthreads();

    const int mtile = wave;
    const int hi8   = (lane >> 4) << 3;
    const int mbase = mtile * 16 + hi8;
    const int n     = lane & 15;

    // ---- ConvTranspose layers via WMMA
    int Lin = 4;
    for (int l = 0; l < A.n_layers; ++l) {
        const _Float16* hin = cur ? bufB : bufA;
        _Float16* hout      = cur ? bufA : bufB;
        const int Lout = Lin * 2;
        const float* bias = A.lb[l];

        // per-lane bias for the 8 output rows this lane owns
        float bias_r[8];
        #pragma unroll
        for (int r = 0; r < 8; ++r) bias_r[r] = bias[mbase + r];

        const int groups = (Lin + 15) >> 4;
        for (int g = 0; g < groups; ++g) {
            const int jn = g * 16 + n;            // lane's central source column
            const bool v1 = jn < Lin;             // col jn
            const bool v0 = (jn >= 1) && (jn <= Lin);   // col jn-1 valid
            const bool v2 = (jn + 1) < Lin;       // col jn+1
            v8f ce = {}, co = {};
            for (int kt = 0; kt < 4; ++kt) {
                const int base_k = kt * 32 + hi8;
                // A fragments (taps 0..3), one v16h global load each
                const v16h* wf = (const v16h*)A.wT +
                                 (size_t)l * 4096 + kt * 256 + mtile * 32 + lane;
                const v16h a0 = wf[0];
                const v16h a1 = wf[1024];
                const v16h a2 = wf[2048];
                const v16h a3 = wf[3072];
                // B fragments (cols jn-1, jn, jn+1), two ds_load_b128 each
                const v16h b0 = load_bfrag(hin, PITCH_A, base_k, jn - 1, Lin, v0);
                const v16h b1 = load_bfrag(hin, PITCH_A, base_k, jn,     Lin, v1);
                const v16h b2 = load_bfrag(hin, PITCH_A, base_k, jn + 1, Lin, v2);
                // even cols: W1*x[j] + W3*x[j-1] ; odd cols: W0*x[j+1] + W2*x[j]
                ce = __builtin_amdgcn_wmma_f32_16x16x32_f16(false, a1, false, b1, (short)0, ce, false, false);
                ce = __builtin_amdgcn_wmma_f32_16x16x32_f16(false, a3, false, b0, (short)0, ce, false, false);
                co = __builtin_amdgcn_wmma_f32_16x16x32_f16(false, a0, false, b2, (short)0, co, false, false);
                co = __builtin_amdgcn_wmma_f32_16x16x32_f16(false, a2, false, b1, (short)0, co, false, false);
            }
            // bias + leaky ReLU(0.2); pack 8 rows -> one ds_store_b128 per parity
            if (v1) {
                v8h pe, po;
                #pragma unroll
                for (int r = 0; r < 8; ++r) {
                    float ve = ce[r] + bias_r[r];
                    float vo = co[r] + bias_r[r];
                    ve = (ve > 0.0f) ? ve : 0.2f * ve;
                    vo = (vo > 0.0f) ? vo : 0.2f * vo;
                    pe[r] = (_Float16)ve;
                    po[r] = (_Float16)vo;
                }
                *(v8h*)(hout + (2 * jn)     * PITCH_A + mbase) = pe;
                *(v8h*)(hout + (2 * jn + 1) * PITCH_A + mbase) = po;
            }
        }
        __syncthreads();
        cur ^= 1;
        Lin = Lout;
    }

    // ---- final conv (1 out ch, k=3, pad=1), input layout [col][ch]
    {
        const _Float16* hin = cur ? bufB : bufA;
        const int nf = A.n_frames;
        if (tid < nf) {
            float acc = A.final_b[0];
            #pragma unroll
            for (int kk = 0; kk < 3; ++kk) {
                const int idx = tid - 1 + kk;
                if (idx >= 0 && idx < nf) {
                    const _Float16* colp = hin + idx * PITCH_A;
                    const float* w3 = A.final_w + kk;
                    for (int c = 0; c < CH; ++c)
                        acc += w3[c * 3] * (float)colp[c];
                }
            }
            A.out[voice * NFRAMES + tid] = acc;
        }
    }
}

// ---------------------------------------------------------------------------
// Post-processing + oscillator bank. One block per voice.
// Uses: cumsum(f_u*pi) = factor * cumsum(f0_u*pi); ha_u[t,h] = env_u[t]*ha[h].
// ---------------------------------------------------------------------------
__global__ __launch_bounds__(256) void synth_kernel(
    const float* __restrict__ seq, const float* __restrict__ amp_p,
    float* __restrict__ out_audio, float* __restrict__ out_f0,
    float* __restrict__ out_env,   float* __restrict__ out_harm)
{
    __shared__ float env_s[NFRAMES], f0_s[NFRAMES], ha_s[NHARM];
    __shared__ float raw[NFRAMES];
    __shared__ float part[256];

    const int tid = threadIdx.x;
    const int v   = blockIdx.x;
    const float amp = amp_p[0];
    const float MIN_F0  = 20.0f / 11025.0f;
    const float F0_DIFF = 8000.0f / 11025.0f - MIN_F0;
    const float PI = 3.14159265358979f;

    // env: nl(seq_env)*amp then reflect box-3
    if (tid < NFRAMES) {
        float s = __sinf(seq[0 * VOICES * NFRAMES + v * NFRAMES + tid]);
        float q = (s + 1.0f) * 0.5f; q *= q;
        raw[tid] = q * amp;
    }
    __syncthreads();
    if (tid < NFRAMES) {
        float a = 0.0f;
        #pragma unroll
        for (int d = -1; d <= 1; ++d) {
            int idx = tid + d;
            if (idx < 0) idx = -idx;
            if (idx > NFRAMES - 1) idx = 2 * (NFRAMES - 1) - idx;
            a += raw[idx];
        }
        env_s[tid] = a * (1.0f / 3.0f);
    }
    __syncthreads();
    // f0: MIN + nl(seq_f0*DIFF) then reflect box-13
    if (tid < NFRAMES) {
        float s = __sinf(seq[1 * VOICES * NFRAMES + v * NFRAMES + tid] * F0_DIFF);
        float q = (s + 1.0f) * 0.5f; q *= q;
        raw[tid] = MIN_F0 + q;
    }
    __syncthreads();
    if (tid < NFRAMES) {
        float a = 0.0f;
        for (int d = -6; d <= 6; ++d) {
            int idx = tid + d;
            if (idx < 0) idx = -idx;
            if (idx > NFRAMES - 1) idx = 2 * (NFRAMES - 1) - idx;
            a += raw[idx];
        }
        f0_s[tid] = a * (1.0f / 13.0f);
    }
    if (tid < NHARM) {
        float s = __sinf(seq[2 * VOICES * NFRAMES + v * NFRAMES + tid]);
        float q = (s + 1.0f) * 0.5f; q *= q;
        ha_s[tid] = q;
    }
    __syncthreads();

    // secondary outputs: f0, env, harm_amp = env (x) ha
    if (tid < NFRAMES) {
        out_f0 [v * NFRAMES + tid] = f0_s[tid];
        out_env[v * NFRAMES + tid] = env_s[tid];
    }
    for (int i = tid; i < NFRAMES * NHARM; i += 256)
        out_harm[v * NFRAMES * NHARM + i] = env_s[i >> 4] * ha_s[i & 15];

    // blocked scan of phase increments, then oscillator bank
    const int CHUNK = NAUDIO / 256;          // 128 samples per thread
    const float scale = (float)NFRAMES / (float)NAUDIO;
    const int t0 = tid * CHUNK;

    float s_local = 0.0f;
    for (int i = 0; i < CHUNK; ++i) {
        const int t = t0 + i;
        float pos = fminf(fmaxf((t + 0.5f) * scale - 0.5f, 0.0f), (float)(NFRAMES - 1));
        const int lo = (int)pos;
        const int hi = min(lo + 1, NFRAMES - 1);
        const float w = pos - (float)lo;
        s_local += (f0_s[lo] + (f0_s[hi] - f0_s[lo]) * w) * PI;
    }
    part[tid] = s_local;
    __syncthreads();
    if (tid == 0) {
        float run = 0.0f;
        for (int i = 0; i < 256; ++i) { float tmp = part[i]; part[i] = run; run += tmp; }
    }
    __syncthreads();

    float phi = part[tid];
    for (int i = 0; i < CHUNK; ++i) {
        const int t = t0 + i;
        float pos = fminf(fmaxf((t + 0.5f) * scale - 0.5f, 0.0f), (float)(NFRAMES - 1));
        const int lo = (int)pos;
        const int hi = min(lo + 1, NFRAMES - 1);
        const float w = pos - (float)lo;
        const float f = f0_s[lo] + (f0_s[hi] - f0_s[lo]) * w;
        const float e = env_s[lo] + (env_s[hi] - env_s[lo]) * w;
        phi += f * PI;                                   // inclusive cumsum
        float acc = __sinf(phi);                         // fundamental
        #pragma unroll
        for (int h = 0; h < NHARM; ++h)
            acc += ha_s[h] * __sinf((float)(h + 2) * phi);
        out_audio[v * NAUDIO + t] = e * acc;
    }
}

// ---------------------------------------------------------------------------
// Host launch
// ---------------------------------------------------------------------------
extern "C" void kernel_launch(void* const* d_in, const int* in_sizes, int n_in,
                              void* d_out, int out_size, void* d_ws, size_t ws_size,
                              hipStream_t stream) {
    (void)n_in; (void)out_size; (void)ws_size;

    struct HostSeq {
        const float *init_w, *init_b, *final_w, *final_b;
        const float *lw[5], *lb[5];
        int n_layers;
    };
    HostSeq hs[3];
    const int nls[3] = { 5, 5, 2 };
    const float* x;
    const float* amp;

    if (in_sizes[0] == 8192) {
        // insertion-order flattening: x, env{...}, f0{...}, ha{...}, amp
        x = (const float*)d_in[0];
        int idx = 1;
        for (int s = 0; s < 3; ++s) {
            hs[s].n_layers = nls[s];
            hs[s].init_w = (const float*)d_in[idx++];
            hs[s].init_b = (const float*)d_in[idx++];
            for (int l = 0; l < nls[s]; ++l) {
                hs[s].lw[l] = (const float*)d_in[idx++];
                hs[s].lb[l] = (const float*)d_in[idx++];
            }
            hs[s].final_w = (const float*)d_in[idx++];
            hs[s].final_b = (const float*)d_in[idx++];
        }
        amp = (const float*)d_in[idx];
    } else {
        // sorted-key (jax pytree) flattening: amp, env{fb,fw,ib,iw,layers}, f0, ha, x
        amp = (const float*)d_in[0];
        int idx = 1;
        for (int s = 0; s < 3; ++s) {
            hs[s].n_layers = nls[s];
            hs[s].final_b = (const float*)d_in[idx++];
            hs[s].final_w = (const float*)d_in[idx++];
            hs[s].init_b  = (const float*)d_in[idx++];
            hs[s].init_w  = (const float*)d_in[idx++];
            for (int l = 0; l < nls[s]; ++l) {
                hs[s].lw[l] = (const float*)d_in[idx++];
                hs[s].lb[l] = (const float*)d_in[idx++];
            }
        }
        x = (const float*)d_in[idx];
    }

    // Workspace: [seq_out: 3*64*128 f32][wT: 12 layers * 65536 f16 fragments]
    float*    seq_out = (float*)d_ws;
    _Float16* wT = (_Float16*)((char*)d_ws + (size_t)3 * VOICES * NFRAMES * sizeof(float));

    // Swizzle all 12 ConvT layer weights into WMMA fragment images
    PrepArgs pa;
    int li = 0;
    for (int s = 0; s < 3; ++s)
        for (int l = 0; l < nls[s]; ++l)
            pa.src[li++] = hs[s].lw[l];
    prep_weights<<<dim3(16, 12), 256, 0, stream>>>(pa, wT);

    // Three sequence decoders (env, f0 @128 frames; ha @16 frames)
    const int nframes[3] = { NFRAMES, NFRAMES, NHARM };
    li = 0;
    for (int s = 0; s < 3; ++s) {
        SeqArgs sa;
        sa.init_w = hs[s].init_w;  sa.init_b = hs[s].init_b;
        for (int l = 0; l < 5; ++l) sa.lb[l] = hs[s].lb[l < nls[s] ? l : 0];
        sa.final_w = hs[s].final_w; sa.final_b = hs[s].final_b;
        sa.wT = wT + (size_t)li * 65536;
        sa.out = seq_out + (size_t)s * VOICES * NFRAMES;
        sa.n_layers = nls[s];
        sa.n_frames = nframes[s];
        seq_kernel<<<VOICES, 256, 0, stream>>>(sa, x);
        li += nls[s];
    }

    // Post + synthesis; outputs concatenated: audio | f0 | env | harm_amp
    float* out = (float*)d_out;
    synth_kernel<<<VOICES, 256, 0, stream>>>(
        seq_out, amp,
        out,
        out + (size_t)VOICES * NAUDIO,
        out + (size_t)VOICES * NAUDIO + (size_t)VOICES * NFRAMES,
        out + (size_t)VOICES * NAUDIO + (size_t)2 * VOICES * NFRAMES);
}
